// TerminalGRU_77979426226918
// MI455X (gfx1250) — matrix operand order repaired and checked by reading, compiled, not run on hardware
//
#include <hip/hip_runtime.h>
#include <hip/hip_bf16.h>
#include <stdint.h>

// Problem constants (match reference)
#define BB 256
#define TT 256
#define DD 512
#define HH 1024
#define CC 128
#define KIN (DD + CC)   // 640
#define G3H (3 * HH)    // 3072

typedef __attribute__((ext_vector_type(16))) __bf16 v16bf;
typedef __attribute__((ext_vector_type(8)))  float  v8f;

union Frag { v16bf v; unsigned u[8]; };
union Acc  { v8f v; float f[8]; };

__device__ __forceinline__ unsigned short f2bf(float x) {
  union { float f; unsigned u; } c; c.f = x;
  unsigned r = c.u + 0x7fffu + ((c.u >> 16) & 1u);   // round-to-nearest-even
  return (unsigned short)(r >> 16);
}
__device__ __forceinline__ unsigned pk2bf(float lo, float hi) {
  return (unsigned)f2bf(lo) | ((unsigned)f2bf(hi) << 16);
}
__device__ __forceinline__ float sigm(float x) { return 1.0f / (1.0f + __expf(-x)); }

// A-fragment K offset for 16-bit 16x32 A layout (ISA 7.12.2):
// lanes 0-15: K {0..7, 16..23}; lanes 16-31: K {8..15, 24..31}; 2 bf16 per VGPR
__device__ __forceinline__ int klocA(int i, int khalf) {
  return khalf + 2 * i + ((i >= 4) ? 8 : 0);
}

__device__ __forceinline__ void load_frag(Frag& f, const unsigned short* row,
                                          int kbase, int khalf) {
#pragma unroll
  for (int i = 0; i < 8; ++i)
    f.u[i] = *(const unsigned*)(row + kbase + klocA(i, khalf));
}
// B fragment: same packing as A, lanes 0-15 K {0..7,16..23}; lanes 16-31 K {8..15,24..31}
__device__ __forceinline__ void load_fragB(Frag& f, const unsigned short* row,
                                           int kbase, int kloB) {
#pragma unroll
  for (int v = 0; v < 8; ++v)
    f.u[v] = *(const unsigned*)(row + kbase + klocA(v, kloB / 2));
}

#define WMMA_BF16(A, B, Cacc) \
  __builtin_amdgcn_wmma_f32_16x16x32_bf16(false, (A), false, (B), (short)0, (Cacc), false, false)

// -------------------- prep: weights -> bf16, zero h0 buffers + counter -----
__global__ void prep_kernel(const float* __restrict__ Wih, const float* __restrict__ Whh,
                            const float* __restrict__ Wout,
                            unsigned short* __restrict__ Wih_b,
                            unsigned short* __restrict__ Whh_b,
                            unsigned short* __restrict__ Wout_b,
                            float* __restrict__ h0f,
                            unsigned* __restrict__ h0_bf,   // BB*HH bf16 as uints
                            unsigned* __restrict__ counter) {
  size_t i = (size_t)blockIdx.x * blockDim.x + threadIdx.x;
  size_t stride = (size_t)gridDim.x * blockDim.x;
  const size_t nih = (size_t)G3H * KIN;
  const size_t nhh = (size_t)G3H * HH;
  const size_t nout = (size_t)CC * HH;
  const size_t nh0 = (size_t)BB * HH;
  for (size_t j = i; j < nih; j += stride) Wih_b[j] = f2bf(Wih[j]);
  for (size_t j = i; j < nhh; j += stride) Whh_b[j] = f2bf(Whh[j]);
  for (size_t j = i; j < nout; j += stride) Wout_b[j] = f2bf(Wout[j]);
  for (size_t j = i; j < nh0; j += stride) h0f[j] = 0.0f;
  for (size_t j = i; j < nh0 / 2; j += stride) h0_bf[j] = 0u;
  if (i == 0) *counter = 0u;
}

// -------------------- prep: inp_bf = bf16([x | shift(true_seq)]) -----------
// One block per GEMM row m = t*B + b; 320 threads, one bf16-pair (uint) each.
__global__ void __launch_bounds__(320)
inp_prep_kernel(const float* __restrict__ x,   // (B,T,D)
                const float* __restrict__ y,   // (B,T,C)
                unsigned* __restrict__ dst) {  // (T*B, 640) bf16 as uint pairs
  const int m = blockIdx.x;
  const int t = m >> 8;
  const int b = m & (BB - 1);
  const int k = threadIdx.x * 2;
  float lo, hi;
  if (k < DD) {
    const float* xr = x + ((size_t)b * TT + t) * DD;
    lo = xr[k]; hi = xr[k + 1];
  } else if (t > 0) {
    const float* yr = y + ((size_t)b * TT + (t - 1)) * CC;
    lo = yr[k - DD]; hi = yr[k - DD + 1];
  } else {
    lo = 0.0f; hi = 0.0f;
  }
  dst[(size_t)m * (KIN / 2) + threadIdx.x] = pk2bf(lo, hi);
}

// -------------------- gates_x = inp_bf @ W_ih^T + b_ih ---------------------
// 2x2 register blocking: each wave computes a 32x32 output tile (4 WMMAs from
// 2 A-frags x 2 B-frags per k-iter -> 16 FLOP/byte of fragment traffic).
// Output layout: gates[(t*B + b) * 3H + g]  (time-major rows for recurrence)
__global__ void __launch_bounds__(256)
gates_gemm_kernel(const unsigned short* __restrict__ inp,  // (T*B, 640) bf16
                  const unsigned short* __restrict__ Wb,   // (3H, 640) bf16
                  const float* __restrict__ bih,           // (3H)
                  float* __restrict__ gates) {             // (T*B, 3H) fp32
  const int lane  = threadIdx.x & 31;
  const int wave  = threadIdx.x >> 5;
  const int m0    = blockIdx.x * 32;                 // 2048 M-blocks of 32 rows
  const int np    = blockIdx.y * 8 + wave;           // 0..95 N-pairs of 32 cols
  const int nrow  = lane & 15;
  const int hi    = lane >> 4;
  const int khalf = hi * 8;
  const int kloB  = hi * 16;
  const int ncol0 = np * 32 + nrow;
  const int ncol1 = ncol0 + 16;

  const unsigned short* arow0 = inp + (size_t)(m0 + nrow) * KIN;
  const unsigned short* arow1 = arow0 + (size_t)16 * KIN;
  const unsigned short* wrow0 = Wb + (size_t)ncol0 * KIN;
  const unsigned short* wrow1 = Wb + (size_t)ncol1 * KIN;

  Acc c00, c01, c10, c11;
#pragma unroll
  for (int i = 0; i < 8; ++i) { c00.f[i] = 0; c01.f[i] = 0; c10.f[i] = 0; c11.f[i] = 0; }

  for (int kk = 0; kk < KIN / 32; ++kk) {
    const int kbase = kk * 32;
    Frag a0, a1, b0, b1;
    load_frag(a0, arow0, kbase, khalf);
    load_frag(a1, arow1, kbase, khalf);
    load_fragB(b0, wrow0, kbase, kloB);
    load_fragB(b1, wrow1, kbase, kloB);
    c00.v = WMMA_BF16(a0.v, b0.v, c00.v);
    c01.v = WMMA_BF16(a0.v, b1.v, c01.v);
    c10.v = WMMA_BF16(a1.v, b0.v, c10.v);
    c11.v = WMMA_BF16(a1.v, b1.v, c11.v);
  }
  const float bias0 = bih[ncol0];
  const float bias1 = bih[ncol1];
#pragma unroll
  for (int i = 0; i < 8; ++i) {
    const int r0 = m0 + i + hi * 8;
    const int r1 = r0 + 16;
    gates[(size_t)r0 * G3H + ncol0] = c00.f[i] + bias0;
    gates[(size_t)r0 * G3H + ncol1] = c01.f[i] + bias1;
    gates[(size_t)r1 * G3H + ncol0] = c10.f[i] + bias0;
    gates[(size_t)r1 * G3H + ncol1] = c11.f[i] + bias1;
  }
}

// -------------------- persistent GRU recurrence over T steps ---------------
// Grid: 256 WGs (16 batch tiles x 16 H tiles of 64 cols), 128 threads (4 waves).
// Step t reads bf16 h from previous step's hs slice (h0_bf for t=0); fp32
// state carried in hA/hB ping-pong for the z*h_prev term. A-frag reused
// across the 3 gate WMMAs.
__global__ void __launch_bounds__(128)
gru_persistent_kernel(const unsigned short* __restrict__ Whh_b, // (3H, H) bf16
                      const float* __restrict__ bhh,            // (3H)
                      const float* __restrict__ gates,          // (T*B, 3H)
                      float* __restrict__ hA, float* __restrict__ hB, // (B,H) fp32, hA==h0==0
                      const unsigned short* __restrict__ h0_bf, // (B,H) bf16 zeros
                      unsigned short* __restrict__ hs,          // (T*B, H) bf16
                      unsigned* __restrict__ counter) {
  const int lane  = threadIdx.x & 31;
  const int wave  = threadIdx.x >> 5;
  const int btile = blockIdx.x & 15;
  const int htile = blockIdx.x >> 4;
  const int b0    = btile * 16;
  const int nrow  = lane & 15;
  const int hi    = lane >> 4;
  const int khalf = hi * 8;
  const int kloB  = hi * 16;
  const int ncol  = htile * 64 + wave * 16 + nrow;   // 0..1023
  const int arow  = b0 + nrow;

  const unsigned short* wr = Whh_b + (size_t)(0 * HH + ncol) * HH;
  const unsigned short* wz = Whh_b + (size_t)(1 * HH + ncol) * HH;
  const unsigned short* wn = Whh_b + (size_t)(2 * HH + ncol) * HH;
  const float bhr = bhh[ncol], bhz = bhh[HH + ncol], bhn = bhh[2 * HH + ncol];
  const unsigned nwg = gridDim.x;

  for (int t = 0; t < TT; ++t) {
    const float* __restrict__ hprev = (t & 1) ? hB : hA;
    float* __restrict__ hnext      = (t & 1) ? hA : hB;
    const unsigned short* hsrc = (t == 0) ? h0_bf : (hs + (size_t)(t - 1) * BB * HH);
    const unsigned short* hrow = hsrc + (size_t)arow * HH;

    Acc ar, az, an;
#pragma unroll
    for (int i = 0; i < 8; ++i) { ar.f[i] = 0.0f; az.f[i] = 0.0f; an.f[i] = 0.0f; }

    for (int kk = 0; kk < HH / 32; ++kk) {
      const int kbase = kk * 32;
      Frag a, br, bz, bn;
      load_frag(a, hrow, kbase, khalf);
      load_fragB(br, wr, kbase, kloB);
      load_fragB(bz, wz, kbase, kloB);
      load_fragB(bn, wn, kbase, kloB);
      ar.v = WMMA_BF16(a.v, br.v, ar.v);
      az.v = WMMA_BF16(a.v, bz.v, az.v);
      an.v = WMMA_BF16(a.v, bn.v, an.v);
    }

#pragma unroll
    for (int i = 0; i < 8; ++i) {
      const int m = b0 + i + hi * 8;                       // batch row
      const size_t grow = ((size_t)t * BB + m) * G3H;
      const float xr = gates[grow + ncol];
      const float xz = gates[grow + HH + ncol];
      const float xn = gates[grow + 2 * HH + ncol];
      const float r  = sigm(xr + ar.f[i] + bhr);
      const float z  = sigm(xz + az.f[i] + bhz);
      const float nn = tanhf(xn + r * (an.f[i] + bhn));
      const float hp = hprev[(size_t)m * HH + ncol];
      const float hv = (1.0f - z) * nn + z * hp;
      hnext[(size_t)m * HH + ncol] = hv;
      hs[((size_t)t * BB + m) * HH + ncol] = f2bf(hv);
    }

    // device-wide barrier: monotonic counter, target = nwg*(t+1)
    __threadfence();
    __syncthreads();
    if (threadIdx.x == 0) {
      __hip_atomic_fetch_add(counter, 1u, __ATOMIC_ACQ_REL, __HIP_MEMORY_SCOPE_AGENT);
      const unsigned target = nwg * (unsigned)(t + 1);
      while (__hip_atomic_load(counter, __ATOMIC_ACQUIRE, __HIP_MEMORY_SCOPE_AGENT) < target)
        __builtin_amdgcn_s_sleep(1);
    }
    __syncthreads();
  }
}

// -------------------- logits = hs @ W_out^T + b_out ------------------------
// 2x1 M-blocking: each wave does a 32x16 tile; one WG (8 waves) covers all
// C=128 columns for 32 M rows.
__global__ void __launch_bounds__(256)
out_gemm_kernel(const unsigned short* __restrict__ hs,   // (T*B, H) bf16
                const unsigned short* __restrict__ Wo,   // (C, H) bf16
                const float* __restrict__ bout,          // (C)
                float* __restrict__ out) {               // (B, T, C) fp32
  const int lane  = threadIdx.x & 31;
  const int wave  = threadIdx.x >> 5;   // ntile 0..7
  const int m0    = blockIdx.x * 32;    // 2048 M-blocks
  const int nrow  = lane & 15;
  const int hi    = lane >> 4;
  const int khalf = hi * 8;
  const int kloB  = hi * 16;
  const int ncol  = wave * 16 + nrow;

  const unsigned short* arow0 = hs + (size_t)(m0 + nrow) * HH;
  const unsigned short* arow1 = arow0 + (size_t)16 * HH;
  const unsigned short* wrow  = Wo + (size_t)ncol * HH;

  Acc c0, c1;
#pragma unroll
  for (int i = 0; i < 8; ++i) { c0.f[i] = 0.0f; c1.f[i] = 0.0f; }

  for (int kk = 0; kk < HH / 32; ++kk) {
    const int kbase = kk * 32;
    Frag a0, a1, bf;
    load_frag(a0, arow0, kbase, khalf);
    load_frag(a1, arow1, kbase, khalf);
    load_fragB(bf, wrow, kbase, kloB);
    c0.v = WMMA_BF16(a0.v, bf.v, c0.v);
    c1.v = WMMA_BF16(a1.v, bf.v, c1.v);
  }
  const float bias = bout[ncol];
#pragma unroll
  for (int i = 0; i < 8; ++i) {
    const int ma = m0 + i + hi * 8;          // m = t*B + b
    const int mb = ma + 16;
    out[((size_t)(ma & (BB - 1)) * TT + (ma >> 8)) * CC + ncol] = c0.f[i] + bias;
    out[((size_t)(mb & (BB - 1)) * TT + (mb >> 8)) * CC + ncol] = c1.f[i] + bias;
  }
}

// ---------------------------------------------------------------------------
extern "C" void kernel_launch(void* const* d_in, const int* in_sizes, int n_in,
                              void* d_out, int out_size, void* d_ws, size_t ws_size,
                              hipStream_t stream) {
  (void)in_sizes; (void)n_in; (void)out_size; (void)ws_size;
  const float* x    = (const float*)d_in[0];
  const float* y    = (const float*)d_in[1];
  const float* Wih  = (const float*)d_in[2];
  const float* bih  = (const float*)d_in[3];
  const float* Whh  = (const float*)d_in[4];
  const float* bhh  = (const float*)d_in[5];
  const float* Wout = (const float*)d_in[6];
  const float* bout = (const float*)d_in[7];
  float* out = (float*)d_out;

  char* ws = (char*)d_ws;
  size_t off = 0;
  auto wsalloc = [&](size_t bytes) -> char* {
    char* p = ws + off;
    off = (off + bytes + 255) & ~(size_t)255;
    return p;
  };
  float*          gates  = (float*)         wsalloc((size_t)TT * BB * G3H * sizeof(float));
  unsigned short* hs     = (unsigned short*)wsalloc((size_t)TT * BB * HH * sizeof(unsigned short));
  unsigned short* inp_bf = (unsigned short*)wsalloc((size_t)TT * BB * KIN * sizeof(unsigned short));
  unsigned short* Wih_b  = (unsigned short*)wsalloc((size_t)G3H * KIN * sizeof(unsigned short));
  unsigned short* Whh_b  = (unsigned short*)wsalloc((size_t)G3H * HH * sizeof(unsigned short));
  unsigned short* Wout_b = (unsigned short*)wsalloc((size_t)CC * HH * sizeof(unsigned short));
  float*          hA     = (float*)         wsalloc((size_t)BB * HH * sizeof(float));
  float*          hB     = (float*)         wsalloc((size_t)BB * HH * sizeof(float));
  unsigned short* h0_bf  = (unsigned short*)wsalloc((size_t)BB * HH * sizeof(unsigned short));
  unsigned*       counter= (unsigned*)      wsalloc(256);

  prep_kernel<<<512, 256, 0, stream>>>(Wih, Whh, Wout, Wih_b, Whh_b, Wout_b,
                                       hA, (unsigned*)h0_bf, counter);

  inp_prep_kernel<<<TT * BB, 320, 0, stream>>>(x, y, (unsigned*)inp_bf);

  dim3 g1((TT * BB) / 32, (G3H / 32) / 8);   // 2048 x 12, 8 waves/WG, 32x32/wave
  gates_gemm_kernel<<<g1, 256, 0, stream>>>(inp_bf, Wih_b, bih, gates);

  gru_persistent_kernel<<<256, 128, 0, stream>>>(Whh_b, bhh, gates, hA, hB,
                                                 h0_bf, hs, counter);

  out_gemm_kernel<<<(TT * BB) / 32, 256, 0, stream>>>(hs, Wout_b, bout, out);
}
